// CLIP_Adapter_1262720385750
// MI455X (gfx1250) — matrix operand order repaired and checked by reading, not compiled, not run on hardware
//
#include <hip/hip_runtime.h>
#include <hip/hip_bf16.h>
#include <math.h>

// ---------------- problem dims ----------------
#define BB 32768
#define DD 512
#define HH 128
#define NN 1000
#define NP 1024   // N padded to multiple of 16

// ---------------- vector types ----------------
typedef __bf16  bf16x16 __attribute__((ext_vector_type(16)));
typedef __bf16  bf16x8  __attribute__((ext_vector_type(8)));
typedef float   f32x8   __attribute__((ext_vector_type(8)));

__device__ __forceinline__ __bf16 f32_to_bf16(float f) {
    union { float f; unsigned u; } in; in.f = f;
    unsigned u = in.u;
    unsigned r = (u + 0x7FFFu + ((u >> 16) & 1u)) >> 16;   // RNE
    union { unsigned short s; __bf16 b; } out; out.s = (unsigned short)r;
    return out.b;
}

__device__ __forceinline__ bf16x16 load_a_frag(const __bf16* base, int lda,
                                               int row, int k0, int halfsel) {
    // 16-bit A layout: lanes 0-15 -> K = k0+[0..7] and k0+16+[0..7]
    //                  lanes 16-31 -> K = k0+8+[0..7] and k0+24+[0..7]
    const __bf16* p = base + (size_t)row * lda + k0 + (halfsel << 3);
    bf16x8 lo = *(const bf16x8*)(p);
    bf16x8 hi = *(const bf16x8*)(p + 16);
    return __builtin_shufflevector(lo, hi, 0,1,2,3,4,5,6,7,8,9,10,11,12,13,14,15);
}

__device__ __forceinline__ bf16x16 load_b_frag(const __bf16* baseT, int ldb,
                                               int col, int k0, int halfsel) {
    // B stored column-major (row = N, contiguous K). lanes 0-15: K=k0+[0..7] and k0+16+[0..7],
    // lanes 16-31: K=k0+8+[0..7] and k0+24+[0..7] -> two 16B loads.
    const __bf16* p = baseT + (size_t)col * ldb + k0 + (halfsel << 3);
    bf16x8 lo = *(const bf16x8*)(p);
    bf16x8 hi = *(const bf16x8*)(p + 16);
    return __builtin_shufflevector(lo, hi, 0,1,2,3,4,5,6,7,8,9,10,11,12,13,14,15);
}

// ---------------- init / finalize ----------------
__global__ void k_init(float* lossAcc, unsigned* accAcc) {
    lossAcc[0] = 0.0f;
    accAcc[0]  = 0u;
}

__global__ void k_finalize(const float* lossAcc, const unsigned* accAcc, float* out) {
    out[0] = lossAcc[0] * (1.0f / (float)BB);
    out[1] = (float)accAcc[0];
}

// ---------------- conversions ----------------
__global__ void k_cvt_img(const float* __restrict__ src, __bf16* __restrict__ dst, int n) {
    int i = blockIdx.x * blockDim.x + threadIdx.x;
    if (i < n) dst[i] = f32_to_bf16(src[i]);
}

// w1 [D,H] -> w1T [H,D] (column-major B for GEMM1)
__global__ void k_prep_w1(const float* __restrict__ w1, __bf16* __restrict__ w1T) {
    int i = blockIdx.x * blockDim.x + threadIdx.x;
    if (i < HH * DD) {
        int n = i >> 9;        // /DD
        int k = i & (DD - 1);
        w1T[i] = f32_to_bf16(w1[k * HH + n]);
    }
}

// w2 [H,D] -> w2T [D,H] (column-major B for GEMM2)
__global__ void k_prep_w2(const float* __restrict__ w2, __bf16* __restrict__ w2T) {
    int i = blockIdx.x * blockDim.x + threadIdx.x;
    if (i < DD * HH) {
        int n = i >> 7;        // /HH
        int k = i & (HH - 1);
        w2T[i] = f32_to_bf16(w2[k * DD + n]);
    }
}

// txt [D,N] -> txtT [NP,D] zero-padded (column-major B for GEMM3)
__global__ void k_prep_txt(const float* __restrict__ txt, __bf16* __restrict__ txtT) {
    int i = blockIdx.x * blockDim.x + threadIdx.x;
    if (i < NP * DD) {
        int n = i >> 9;        // /DD
        int k = i & (DD - 1);
        txtT[i] = (n < NN) ? f32_to_bf16(txt[k * NN + n]) : f32_to_bf16(0.0f);
    }
}

// ---------------- GEMM1: h = relu(img @ w1 + b1), [B,H] ----------------
__global__ void k_gemm1(const __bf16* __restrict__ A, const __bf16* __restrict__ Bt,
                        const float* __restrict__ b1, __bf16* __restrict__ Hout) {
    const int mtile   = blockIdx.x * 16;
    const int wave    = threadIdx.x >> 5;
    const int lane    = threadIdx.x & 31;
    const int halfsel = lane >> 4;
    const int l16     = lane & 15;
    const int n0      = wave * 16;          // 8 waves cover H=128
    const int arow    = mtile + l16;
    const int bcol    = n0 + l16;

    f32x8 acc = {};
    #pragma unroll
    for (int k0 = 0; k0 < DD; k0 += 32) {
        bf16x16 af = load_a_frag(A, DD, arow, k0, halfsel);
        bf16x16 bf = load_b_frag(Bt, DD, bcol, k0, halfsel);
        acc = __builtin_amdgcn_wmma_f32_16x16x32_bf16(false, af, false, bf,
                                                      (short)0, acc, false, false);
    }
    const int   col  = n0 + l16;
    const float bias = b1[col];
    #pragma unroll
    for (int j = 0; j < 8; ++j) {
        int   row = mtile + j + halfsel * 8;
        float v   = acc[j] + bias;
        v = v > 0.0f ? v : 0.0f;
        Hout[(size_t)row * HH + col] = f32_to_bf16(v);
    }
}

// -------- GEMM2: adapted = relu(h @ w2 + b2); blend with img -> bf16 --------
__global__ void k_gemm2(const __bf16* __restrict__ A, const __bf16* __restrict__ Bt,
                        const float* __restrict__ b2, const float* __restrict__ img,
                        const float* __restrict__ alpha_p, __bf16* __restrict__ Out) {
    const int mtile   = blockIdx.x * 16;
    const int ngrp    = blockIdx.y;         // 0..3, 128 cols each
    const int wave    = threadIdx.x >> 5;
    const int lane    = threadIdx.x & 31;
    const int halfsel = lane >> 4;
    const int l16     = lane & 15;
    const int n0      = ngrp * 128 + wave * 16;
    const int arow    = mtile + l16;
    const int bcol    = n0 + l16;

    f32x8 acc = {};
    #pragma unroll
    for (int k0 = 0; k0 < HH; k0 += 32) {
        bf16x16 af = load_a_frag(A, HH, arow, k0, halfsel);
        bf16x16 bf = load_b_frag(Bt, HH, bcol, k0, halfsel);
        acc = __builtin_amdgcn_wmma_f32_16x16x32_bf16(false, af, false, bf,
                                                      (short)0, acc, false, false);
    }
    const int   col  = n0 + l16;
    const float bias = b2[col];
    const float a    = alpha_p[0];
    #pragma unroll
    for (int j = 0; j < 8; ++j) {
        int   row = mtile + j + halfsel * 8;
        float v   = acc[j] + bias;
        v = v > 0.0f ? v : 0.0f;                       // relu(adapted)
        float blended = a * img[(size_t)row * DD + col] + (1.0f - a) * v;
        Out[(size_t)row * DD + col] = f32_to_bf16(blended);
    }
}

// -------- GEMM3 + fused row reductions (norm, log-softmax NLL, argmax) --------
__global__ void k_gemm3_reduce(const __bf16* __restrict__ A, const __bf16* __restrict__ Tt,
                               const int* __restrict__ tgt, const int* __restrict__ t_p,
                               float* lossAcc, unsigned* accAcc) {
    extern __shared__ float smem[];
    float* SIM  = smem;                 // 16 x 1024
    float* PMAX = smem + 16 * NP;       // 256
    float* PIDX = PMAX + 256;           // 256
    float* PSS  = PIDX + 256;           // 256
    float* PSE  = PSS + 256;            // 256
    float* ROWS = PSE + 256;            // 32

    const int mtile   = blockIdx.x * 16;
    const int wave    = threadIdx.x >> 5;
    const int lane    = threadIdx.x & 31;
    const int halfsel = lane >> 4;
    const int l16     = lane & 15;
    const int arow    = mtile + l16;

    // ---- phase A: 8 waves x 8 tiles = 64 N-tiles (N=1024), K=512 ----
    f32x8 accs[8];
    #pragma unroll
    for (int i = 0; i < 8; ++i) accs[i] = (f32x8){};

    for (int k0 = 0; k0 < DD; k0 += 32) {
        bf16x16 af = load_a_frag(A, DD, arow, k0, halfsel);
        if (k0 + 32 < DD)
            __builtin_prefetch(A + (size_t)arow * DD + k0 + 32, 0, 1);
        #pragma unroll
        for (int i = 0; i < 8; ++i) {
            int bcol = (wave * 8 + i) * 16 + l16;
            bf16x16 bf = load_b_frag(Tt, DD, bcol, k0, halfsel);
            accs[i] = __builtin_amdgcn_wmma_f32_16x16x32_bf16(false, af, false, bf,
                                                              (short)0, accs[i],
                                                              false, false);
        }
    }
    // spill sim tile to LDS
    #pragma unroll
    for (int i = 0; i < 8; ++i) {
        int ncol = (wave * 8 + i) * 16 + l16;
        #pragma unroll
        for (int j = 0; j < 8; ++j) {
            int row = j + halfsel * 8;
            SIM[row * NP + ncol] = accs[i][j];
        }
    }
    __syncthreads();

    // ---- phase B: per-row reductions (16 threads per row) ----
    const int tid = threadIdx.x;
    const int row = tid >> 4;
    const int sub = tid & 15;

    float mx = -3.4e38f, bid = 1e9f, ss = 0.0f;
    for (int c = sub; c < NN; c += 16) {
        float v = SIM[row * NP + c];
        ss += v * v;
        if (v > mx) { mx = v; bid = (float)c; }
    }
    PMAX[tid] = mx; PIDX[tid] = bid; PSS[tid] = ss;
    __syncthreads();

    if (sub == 0) {
        float m = -3.4e38f, id = 1e9f, s = 0.0f;
        for (int q = 0; q < 16; ++q) {
            float pm = PMAX[row * 16 + q];
            float pi = PIDX[row * 16 + q];
            s += PSS[row * 16 + q];
            if (pm > m)            { m = pm; id = pi; }
            else if (pm == m && pi < id) { id = pi; }
        }
        float tf   = (float)t_p[0];
        float norm = sqrtf(s);                 // exp(logit_scale) cancels in normalization
        float inv  = 1.0f / (norm * tf);
        ROWS[row * 2 + 0] = inv;
        ROWS[row * 2 + 1] = m * inv;           // max of z = sim*inv
        PIDX[row * 16]    = id;                // stash argmax
    }
    __syncthreads();

    float inv = ROWS[row * 2 + 0];
    float mz  = ROWS[row * 2 + 1];
    float se  = 0.0f;
    for (int c = sub; c < NN; c += 16) {
        float z = SIM[row * NP + c] * inv;
        se += __expf(z - mz);
    }
    PSE[tid] = se;
    __syncthreads();

    if (sub == 0) {
        float tot = 0.0f;
        for (int q = 0; q < 16; ++q) tot += PSE[row * 16 + q];
        int   grow = mtile + row;
        int   tg   = tgt[grow];
        float zt   = SIM[row * NP + tg] * inv;
        float lossRow = -(zt - mz - logf(tot));
        atomicAdd(lossAcc, lossRow);
        int am = (int)PIDX[row * 16];
        if (am == tg) atomicAdd(accAcc, 1u);
    }
}

// ---------------- host-side launcher ----------------
extern "C" void kernel_launch(void* const* d_in, const int* in_sizes, int n_in,
                              void* d_out, int out_size, void* d_ws, size_t ws_size,
                              hipStream_t stream) {
    const float* img   = (const float*)d_in[0];   // [B,D]
    const float* txt   = (const float*)d_in[1];   // [D,N]
    const float* w1    = (const float*)d_in[2];   // [D,H]
    const float* b1    = (const float*)d_in[3];   // [H]
    const float* w2    = (const float*)d_in[4];   // [H,D]
    const float* b2    = (const float*)d_in[5];   // [D]
    const float* alpha = (const float*)d_in[6];   // scalar
    // d_in[7] = logit_scale: cancels in row normalization, unused
    const int*   tgt   = (const int*)d_in[8];     // [B]
    const int*   t_p   = (const int*)d_in[9];     // scalar

    char* ws = (char*)d_ws;
    float*    lossAcc = (float*)ws;
    unsigned* accAcc  = (unsigned*)(ws + 8);
    __bf16* imgbf = (__bf16*)(ws + 256);
    __bf16* hbf   = imgbf + (size_t)BB * DD;
    __bf16* adbf  = hbf   + (size_t)BB * HH;
    __bf16* w1T   = adbf  + (size_t)BB * DD;
    __bf16* w2T   = w1T   + (size_t)HH * DD;
    __bf16* txtT  = w2T   + (size_t)DD * HH;

    k_init<<<1, 1, 0, stream>>>(lossAcc, accAcc);

    k_cvt_img<<<(BB * DD + 255) / 256, 256, 0, stream>>>(img, imgbf, BB * DD);
    k_prep_w1<<<(HH * DD + 255) / 256, 256, 0, stream>>>(w1, w1T);
    k_prep_w2<<<(DD * HH + 255) / 256, 256, 0, stream>>>(w2, w2T);
    k_prep_txt<<<(NP * DD + 255) / 256, 256, 0, stream>>>(txt, txtT);

    k_gemm1<<<BB / 16, 256, 0, stream>>>(imgbf, w1T, b1, hbf);
    k_gemm2<<<dim3(BB / 16, 4), 256, 0, stream>>>(hbf, w2T, b2, img, alpha, adbf);

    size_t shbytes = (size_t)(16 * NP + 4 * 256 + 32) * sizeof(float);  // ~70 KB (<320 KB WGP LDS)
    k_gemm3_reduce<<<BB / 16, 256, shbytes, stream>>>(adbf, txtT, tgt, t_p, lossAcc, accAcc);

    k_finalize<<<1, 1, 0, stream>>>(lossAcc, accAcc, (float*)d_out);
}
